// RNNTopic_69114613728640
// MI455X (gfx1250) — matrix operand order repaired and checked by reading, not compiled, not run on hardware
//
#include <hip/hip_runtime.h>
#include <hip/hip_bf16.h>

typedef __attribute__((ext_vector_type(16))) __bf16 v16bf;
typedef __attribute__((ext_vector_type(4)))  __bf16 v4bf;
typedef __attribute__((ext_vector_type(8)))  float  v8f;
typedef __attribute__((ext_vector_type(4)))  float  v4f;
typedef unsigned short u16b;
typedef unsigned int   u32b;

#define B_    128
#define T_    256
#define E_    512
#define H_    1024
#define GH_   4096
#define TOP_  128
#define F_    512
#define O_    10
#define KCAT_ 1536          // E_ + H_
#define LDA_  40            // u16 stride for a 32-wide K row (+8 pad vs banks)

union FragBF  { v16bf v; uint4 q[2]; };
union Pack16  { u16b u[16]; v4bf b4[4]; uint4 q[2]; };

__device__ __forceinline__ u16b f2bf(float f) {
  u32b u = __builtin_bit_cast(u32b, f);
  u32b r = u + 0x7FFFu + ((u >> 16) & 1u);   // round-to-nearest-even
  return (u16b)(r >> 16);
}
// HW packed convert path (v_cvt_pk_bf16_f32 on gfx1250), RNE.
__device__ __forceinline__ v4bf cvt4(float4 f) {
  v4f vf = {f.x, f.y, f.z, f.w};
  return __builtin_convertvector(vf, v4bf);
}
__device__ __forceinline__ float sigmoidf_(float x) { return 1.0f / (1.0f + __expf(-x)); }

// CDNA5 async copy: 16B/lane global -> LDS, tracked by ASYNCcnt (no VGPR data).
__device__ __forceinline__ void async_b128(unsigned lds_off, const void* gsrc) {
  asm volatile("global_load_async_to_lds_b128 %0, %1, off"
               :: "v"(lds_off), "v"(gsrc) : "memory");
}
__device__ __forceinline__ void wait_async0() {
  asm volatile("s_wait_asynccnt 0x0" ::: "memory");
}

// ---------------------------------------------------------------------------
// Prep: W_cat = bf16([w_ih | w_hh]) (4096 x 1536), fc1_w -> bf16, zero h0/c0
// ---------------------------------------------------------------------------
__global__ void prep_kernel(const float* __restrict__ w_ih, const float* __restrict__ w_hh,
                            const float* __restrict__ fc1_w,
                            u16b* __restrict__ wcat, u16b* __restrict__ fc1wb,
                            float* __restrict__ hx0, float* __restrict__ cx0) {
  int n = blockIdx.x * blockDim.x + threadIdx.x;
  if (n < GH_ * KCAT_) {
    int row = n / KCAT_, k = n - row * KCAT_;
    float v = (k < E_) ? w_ih[row * E_ + k] : w_hh[row * H_ + (k - E_)];
    wcat[n] = f2bf(v);
  }
  if (n < F_ * H_) fc1wb[n] = f2bf(fc1_w[n]);
  if (n < B_ * H_) { hx0[n] = 0.0f; cx0[n] = 0.0f; }
}

// ---------------------------------------------------------------------------
// tbias[b, 4H] = x_top @ w_th^T + b   (one-time, kept fp32 for accuracy)
// ---------------------------------------------------------------------------
__global__ void tbias_kernel(const float* __restrict__ x_top, const float* __restrict__ w_th,
                             const float* __restrict__ bias, float* __restrict__ tbias) {
  int n = blockIdx.x * blockDim.x + threadIdx.x;          // B_*GH_ threads
  int b = n >> 12, col = n & (GH_ - 1);
  const float* xt = x_top + b * TOP_;
  const float* wt = w_th + (size_t)col * TOP_;
  float acc = bias[col];
#pragma unroll 4
  for (int k = 0; k < TOP_; ++k) acc += xt[k] * wt[k];
  tbias[n] = acc;
}

// ---------------------------------------------------------------------------
// One LSTM time step, fully fused:
//   gates[128,4096] = [emb[x[:,t]] | h_prev](bf16) @ W_cat^T(bf16) + tbias
//   pointwise i/f/g/o -> c_new, h_new
// Block = 256 thr (8 waves). Block n owns cols {g*1024 + n0 .. n0+31}, g=0..3,
// so all four gates for its slice are local -> pointwise fused in-kernel.
// B tiles (bf16 weights) are staged with GLOBAL_LOAD_ASYNC_TO_LDS_B128;
// A tiles (fp32 gather + packed cvt) go through VALU.
// ---------------------------------------------------------------------------
__global__ void __launch_bounds__(256)
lstm_step_kernel(const int* __restrict__ x, const float* __restrict__ emb,
                 const u16b* __restrict__ wcat, const float* __restrict__ tbias,
                 const float* __restrict__ hx_in, const float* __restrict__ cx_in,
                 float* __restrict__ hx_out, float* __restrict__ cx_out, int t) {
  __shared__ int idx_s[B_];
  __shared__ __align__(16) unsigned char smem[B_ * 132 * 4];       // 67584 B, aliased
  u16b*  ldsA = (u16b*)smem;                                       // [2][128*LDA_]
  u16b*  ldsB = (u16b*)(smem + 2 * B_ * LDA_ * sizeof(u16b));      // [2][128*LDA_]
  float* ldsG = (float*)smem;                                      // [128][132] (after K loop)

  const int tid = threadIdx.x;
  const int lid = tid & 31, w = tid >> 5;
  const int mw = w >> 2, g = w & 3;        // 2 M-waves x 4 gate-waves
  const int n0 = blockIdx.x * 32;          // 32 blocks cover H_=1024 cols per gate
  const int srow = tid >> 1, shalf = tid & 1;
  const int arow = lid & 15, ahalf = lid >> 4;

  if (tid < B_) idx_s[tid] = x[tid * T_ + t];

  {  // warm epilogue operands into cache while the GEMM runs
    int prow = tid >> 1;
    int pcol = n0 + ((tid & 1) << 4);
    __builtin_prefetch(tbias + (size_t)prow * GH_ + pcol, 0, 3);
    __builtin_prefetch(tbias + (size_t)prow * GH_ + H_ + pcol, 0, 3);
    __builtin_prefetch(tbias + (size_t)prow * GH_ + 2 * H_ + pcol, 0, 3);
    __builtin_prefetch(tbias + (size_t)prow * GH_ + 3 * H_ + pcol, 0, 3);
    __builtin_prefetch(cx_in + (size_t)prow * H_ + pcol, 0, 3);
  }
  __syncthreads();

  v8f acc[4][2] = {};
  const int KT = KCAT_ / 32;               // 48 K-chunks

  // A: gather embedding rows (K<512) or h_prev (K>=512), fp32 -> bf16 -> LDS
  auto stageA = [&](int kt, int buf) {
    const int kglob = kt * 32;
    const float* asrc = (kglob < E_)
        ? (emb + (size_t)idx_s[srow] * E_ + kglob + shalf * 16)
        : (hx_in + (size_t)srow * H_ + (kglob - E_) + shalf * 16);
    float4 f0 = ((const float4*)asrc)[0];
    float4 f1 = ((const float4*)asrc)[1];
    float4 f2 = ((const float4*)asrc)[2];
    float4 f3 = ((const float4*)asrc)[3];
    Pack16 pk;
    pk.b4[0] = cvt4(f0); pk.b4[1] = cvt4(f1); pk.b4[2] = cvt4(f2); pk.b4[3] = cvt4(f3);
    uint4* adst = (uint4*)&ldsA[buf * (B_ * LDA_) + srow * LDA_ + shalf * 16];
    adst[0] = pk.q[0]; adst[1] = pk.q[1];
  };

  // B: bf16 weight rows, async DMA straight into LDS (2 x b128 per thread).
  // LDS row r <-> global gate row (r>>5)*1024 + n0 + (r&31)
  auto stageB = [&](int kt, int buf) {
    const int kglob = kt * 32;
#pragma unroll
    for (int r = 0; r < 2; ++r) {
      int unit = tid + r * 256;            // 512 units of 16B = 8KB chunk
      int row = unit >> 2, seg = unit & 3;
      int grow = (row >> 5) * H_ + n0 + (row & 31);
      const u16b* gsrc = wcat + (size_t)grow * KCAT_ + kglob + seg * 8;
      unsigned lds_off =
          (unsigned)(size_t)&ldsB[buf * (B_ * LDA_) + row * LDA_ + seg * 8];
      async_b128(lds_off, gsrc);
    }
  };

  auto compute = [&](int buf) {
    FragBF a[4], bb[2];
#pragma unroll
    for (int i = 0; i < 4; ++i) {   // A 16x32 bf16 tiles: lane<16 K{0..15}, lane>=16 K{16..31}
      const u16b* p = &ldsA[buf * (B_ * LDA_) + (mw * 64 + i * 16 + arow) * LDA_ + ahalf * 16];
      a[i].q[0] = *(const uint4*)p;
      a[i].q[1] = *(const uint4*)(p + 8);
    }
#pragma unroll
    for (int j = 0; j < 2; ++j) {   // B 32x16 bf16 tiles: lane = col, K half by lane>=16
      const u16b* p = &ldsB[buf * (B_ * LDA_) + (g * 32 + j * 16 + arow) * LDA_ + ahalf * 16];
      bb[j].q[0] = *(const uint4*)p;
      bb[j].q[1] = *(const uint4*)(p + 8);
    }
#pragma unroll
    for (int i = 0; i < 4; ++i)
#pragma unroll
      for (int j = 0; j < 2; ++j)
        acc[i][j] = __builtin_amdgcn_wmma_f32_16x16x32_bf16(
            false, a[i].v, false, bb[j].v, (short)0, acc[i][j], false, false);
  };

  stageB(0, 0);
  stageA(0, 0);
  wait_async0();
  __syncthreads();
  for (int kt = 0; kt < KT; ++kt) {
    if (kt + 1 < KT) {                     // fill the other buffer
      stageB(kt + 1, (kt + 1) & 1);        // async DMA overlaps the WMMAs below
      stageA(kt + 1, (kt + 1) & 1);
    }
    compute(kt & 1);
    wait_async0();                         // B chunk landed in LDS
    __syncthreads();
  }

  // Spill accumulators to LDS so each thread can see all four gates of its cols.
#pragma unroll
  for (int i = 0; i < 4; ++i)
#pragma unroll
    for (int j = 0; j < 2; ++j) {
      int rowb = mw * 64 + i * 16 + (lid >> 4) * 8;   // C/D layout: lanes>=16 hold M+8
      int col  = g * 32 + j * 16 + (lid & 15);
#pragma unroll
      for (int r = 0; r < 8; ++r)
        ldsG[(rowb + r) * 132 + col] = acc[i][j][r];
    }
  __syncthreads();

  // Fused LSTM pointwise on the block's 128 x 32 slice.
  for (int e = tid; e < B_ * 32; e += 256) {
    int row = e >> 5, c = e & 31;
    int col = n0 + c;
    float iv = ldsG[row * 132 +  0 + c] + tbias[(size_t)row * GH_ +          col];
    float fv = ldsG[row * 132 + 32 + c] + tbias[(size_t)row * GH_ + H_     + col];
    float gv = ldsG[row * 132 + 64 + c] + tbias[(size_t)row * GH_ + 2 * H_ + col];
    float ov = ldsG[row * 132 + 96 + c] + tbias[(size_t)row * GH_ + 3 * H_ + col];
    float ig = sigmoidf_(iv), fg = sigmoidf_(fv), og = sigmoidf_(ov);
    float cn = fg * cx_in[row * H_ + col] + ig * tanhf(gv);
    float hn = og * tanhf(cn);
    cx_out[row * H_ + col] = cn;
    hx_out[row * H_ + col] = hn;
  }
}

// ---------------------------------------------------------------------------
// fc1: h1[128,512] = h_final @ fc1_w^T + fc1_b  (WMMA bf16, K=1024)
// ---------------------------------------------------------------------------
__global__ void __launch_bounds__(256)
fc1_kernel(const float* __restrict__ hx, const u16b* __restrict__ fc1wb,
           const float* __restrict__ fc1_b, float* __restrict__ h1) {
  __shared__ __align__(16) u16b ldsA[2][B_ * LDA_];
  __shared__ __align__(16) u16b ldsB[2][B_ * LDA_];
  const int tid = threadIdx.x, lid = tid & 31, w = tid >> 5;
  const int mw = w >> 2, nw = w & 3;
  const int n0 = blockIdx.x * 128;         // 4 blocks cover F_=512
  const int srow = tid >> 1, shalf = tid & 1;
  const int arow = lid & 15, ahalf = lid >> 4;

  v8f acc[4][2] = {};
  const int KT = H_ / 32;                  // 32 K-chunks

  auto stageA = [&](int kt, int buf) {
    const int kglob = kt * 32;
    const float* asrc = hx + (size_t)srow * H_ + kglob + shalf * 16;
    float4 f0 = ((const float4*)asrc)[0];
    float4 f1 = ((const float4*)asrc)[1];
    float4 f2 = ((const float4*)asrc)[2];
    float4 f3 = ((const float4*)asrc)[3];
    Pack16 pk;
    pk.b4[0] = cvt4(f0); pk.b4[1] = cvt4(f1); pk.b4[2] = cvt4(f2); pk.b4[3] = cvt4(f3);
    uint4* adst = (uint4*)&ldsA[buf][srow * LDA_ + shalf * 16];
    adst[0] = pk.q[0]; adst[1] = pk.q[1];
  };

  auto stageB = [&](int kt, int buf) {
    const int kglob = kt * 32;
#pragma unroll
    for (int r = 0; r < 2; ++r) {
      int unit = tid + r * 256;
      int row = unit >> 2, seg = unit & 3;
      const u16b* gsrc = fc1wb + (size_t)(n0 + row) * H_ + kglob + seg * 8;
      unsigned lds_off = (unsigned)(size_t)&ldsB[buf][row * LDA_ + seg * 8];
      async_b128(lds_off, gsrc);
    }
  };

  auto compute = [&](int buf) {
    FragBF a[4], bb[2];
#pragma unroll
    for (int i = 0; i < 4; ++i) {
      const u16b* p = &ldsA[buf][(mw * 64 + i * 16 + arow) * LDA_ + ahalf * 16];
      a[i].q[0] = *(const uint4*)p;
      a[i].q[1] = *(const uint4*)(p + 8);
    }
#pragma unroll
    for (int j = 0; j < 2; ++j) {
      const u16b* p = &ldsB[buf][(nw * 32 + j * 16 + arow) * LDA_ + ahalf * 16];
      bb[j].q[0] = *(const uint4*)p;
      bb[j].q[1] = *(const uint4*)(p + 8);
    }
#pragma unroll
    for (int i = 0; i < 4; ++i)
#pragma unroll
      for (int j = 0; j < 2; ++j)
        acc[i][j] = __builtin_amdgcn_wmma_f32_16x16x32_bf16(
            false, a[i].v, false, bb[j].v, (short)0, acc[i][j], false, false);
  };

  stageB(0, 0);
  stageA(0, 0);
  wait_async0();
  __syncthreads();
  for (int kt = 0; kt < KT; ++kt) {
    if (kt + 1 < KT) {
      stageB(kt + 1, (kt + 1) & 1);
      stageA(kt + 1, (kt + 1) & 1);
    }
    compute(kt & 1);
    wait_async0();
    __syncthreads();
  }

#pragma unroll
  for (int i = 0; i < 4; ++i)
#pragma unroll
    for (int j = 0; j < 2; ++j) {
      int rowb = mw * 64 + i * 16 + (lid >> 4) * 8;
      int col  = n0 + nw * 32 + j * 16 + (lid & 15);
      float bias = fc1_b[col];
#pragma unroll
      for (int r = 0; r < 8; ++r)
        h1[(size_t)(rowb + r) * F_ + col] = acc[i][j][r] + bias;
    }
}

// ---------------------------------------------------------------------------
// BatchNorm1d over batch (training stats), per feature.
// ---------------------------------------------------------------------------
__global__ void bn_kernel(const float* __restrict__ h1, const float* __restrict__ gamma,
                          const float* __restrict__ beta, float* __restrict__ h1n) {
  int f = threadIdx.x;                    // 512 threads
  float s = 0.0f, ss = 0.0f;
  for (int b = 0; b < B_; ++b) { float v = h1[b * F_ + f]; s += v; ss += v * v; }
  float mu  = s * (1.0f / B_);
  float var = ss * (1.0f / B_) - mu * mu;
  float sc  = rsqrtf(var + 1e-5f) * gamma[f];
  float bt  = beta[f];
  for (int b = 0; b < B_; ++b) h1n[b * F_ + f] = (h1[b * F_ + f] - mu) * sc + bt;
}

// ---------------------------------------------------------------------------
// fc2: out[128,10] = h1n @ fc2_w^T + fc2_b  (tiny; wave-reduced dot)
// ---------------------------------------------------------------------------
__global__ void fc2_kernel(const float* __restrict__ h1n, const float* __restrict__ fc2_w,
                           const float* __restrict__ fc2_b, float* __restrict__ out) {
  int b = blockIdx.x, l = threadIdx.x;    // 32 threads = 1 wave
  const float* hrow = h1n + (size_t)b * F_;
  for (int o = 0; o < O_; ++o) {
    const float* wrow = fc2_w + (size_t)o * F_;
    float p = 0.0f;
    for (int f = l; f < F_; f += 32) p += hrow[f] * wrow[f];
#pragma unroll
    for (int m = 16; m > 0; m >>= 1) p += __shfl_xor(p, m, 32);
    if (l == 0) out[b * O_ + o] = p + fc2_b[o];
  }
}

// ---------------------------------------------------------------------------
extern "C" void kernel_launch(void* const* d_in, const int* in_sizes, int n_in,
                              void* d_out, int out_size, void* d_ws, size_t ws_size,
                              hipStream_t stream) {
  (void)in_sizes; (void)n_in; (void)out_size; (void)ws_size;
  const int*   x     = (const int*)  d_in[0];
  const float* x_top = (const float*)d_in[1];
  const float* emb   = (const float*)d_in[2];
  const float* w_ih  = (const float*)d_in[3];
  const float* w_hh  = (const float*)d_in[4];
  const float* w_th  = (const float*)d_in[5];
  const float* bias  = (const float*)d_in[6];
  const float* fc1_w = (const float*)d_in[7];
  const float* fc1_b = (const float*)d_in[8];
  const float* gamma = (const float*)d_in[9];
  const float* betap = (const float*)d_in[10];
  const float* fc2_w = (const float*)d_in[11];
  const float* fc2_b = (const float*)d_in[12];
  float* out = (float*)d_out;

  char* p = (char*)d_ws;
  auto alloc = [&](size_t bytes) {
    char* r = p; p += (bytes + 255) & ~(size_t)255; return r;
  };
  u16b*  wcat  = (u16b*)alloc((size_t)GH_ * KCAT_ * 2);   // 12.6 MB, L2-resident
  u16b*  fc1wb = (u16b*)alloc((size_t)F_ * H_ * 2);
  float* tb    = (float*)alloc((size_t)B_ * GH_ * 4);
  float* hx0   = (float*)alloc((size_t)B_ * H_ * 4);
  float* hx1   = (float*)alloc((size_t)B_ * H_ * 4);
  float* cx0   = (float*)alloc((size_t)B_ * H_ * 4);
  float* cx1   = (float*)alloc((size_t)B_ * H_ * 4);
  float* h1    = (float*)alloc((size_t)B_ * F_ * 4);
  float* h1n   = (float*)alloc((size_t)B_ * F_ * 4);
  float* hxb[2] = { hx0, hx1 };
  float* cxb[2] = { cx0, cx1 };

  int prepN = GH_ * KCAT_;
  prep_kernel<<<(prepN + 255) / 256, 256, 0, stream>>>(w_ih, w_hh, fc1_w, wcat, fc1wb, hx0, cx0);
  tbias_kernel<<<(B_ * GH_) / 256, 256, 0, stream>>>(x_top, w_th, bias, tb);

  for (int t = 0; t < T_; ++t)
    lstm_step_kernel<<<32, 256, 0, stream>>>(x, emb, wcat, tb,
                                             hxb[t & 1], cxb[t & 1],
                                             hxb[(t + 1) & 1], cxb[(t + 1) & 1], t);

  // T_ = 256 (even) -> final state landed back in buffer 0.
  fc1_kernel<<<F_ / 128, 256, 0, stream>>>(hxb[0], fc1wb, fc1_b, h1);
  bn_kernel<<<1, F_, 0, stream>>>(h1, gamma, betap, h1n);
  fc2_kernel<<<B_, 32, 0, stream>>>(h1n, fc2_w, fc2_b, out);
}
